// Block_53068615909973
// MI455X (gfx1250) — matrix operand order repaired and checked by reading, not compiled, not run on hardware
//
#include <hip/hip_runtime.h>
#include <math.h>

// ---------------- problem constants ----------------
constexpr int B   = 2;
constexpr int T   = 1024;
constexpr int C   = 2048;
constexpr int HS  = 64;
constexpr int H   = C / HS;     // 32
constexpr int FFN = 7168;
constexpr int TM  = 32;         // 5*TM = 160
constexpr int TD  = 64;
constexpr int BT  = B * T;      // 2048 rows
constexpr int BTC = BT * C;     // 4,194,304

typedef __attribute__((ext_vector_type(16))) __bf16 bf16x16;
typedef __attribute__((ext_vector_type(8)))  float  f32x8;
typedef __attribute__((ext_vector_type(4)))  int    v4i;

// ---- CDNA5 async global->LDS path (guarded; falls back to reg-staged copies) ----
#if defined(__AMDGCN__) && __has_builtin(__builtin_amdgcn_global_load_async_to_lds_b128)
#define HAVE_ASYNC_LDS 1
#else
#define HAVE_ASYNC_LDS 0
#endif

#if defined(__AMDGCN__) && __has_builtin(__builtin_amdgcn_s_wait_asynccnt)
#define WAIT_ASYNC0() __builtin_amdgcn_s_wait_asynccnt(0)
#elif defined(__AMDGCN__)
#define WAIT_ASYNC0() asm volatile("s_wait_asynccnt 0x0" ::: "memory")
#else
#define WAIT_ASYNC0()
#endif

// =====================================================================
// fp32 -> bf16 conversion
// =====================================================================
__global__ __launch_bounds__(256) void cvt_bf16_kernel(const float* __restrict__ s,
                                                       __bf16* __restrict__ d, int n) {
  int i = blockIdx.x * 256 + threadIdx.x;
  if (i < n) d[i] = (__bf16)s[i];
}

// =====================================================================
// LayerNorm over C per token.  block = 256 threads, 8 elems/thread.
// =====================================================================
__global__ __launch_bounds__(256) void ln_kernel(const float* __restrict__ x,
                                                 const float* __restrict__ w,
                                                 const float* __restrict__ b,
                                                 float* __restrict__ out) {
  int row = blockIdx.x;
  const float* xr = x + (size_t)row * C;
  float* yr = out + (size_t)row * C;
  float v[8];
  float s = 0.f, s2 = 0.f;
#pragma unroll
  for (int i = 0; i < 8; i++) {
    int c = threadIdx.x + i * 256;
    v[i] = xr[c];
    s += v[i];
    s2 += v[i] * v[i];
  }
#pragma unroll
  for (int off = 16; off > 0; off >>= 1) {
    s  += __shfl_down(s, off, 32);
    s2 += __shfl_down(s2, off, 32);
  }
  __shared__ float ps[8], ps2[8];
  int wave = threadIdx.x >> 5, lane = threadIdx.x & 31;
  if (lane == 0) { ps[wave] = s; ps2[wave] = s2; }
  __syncthreads();
  if (threadIdx.x == 0) {
    float a = 0.f, a2 = 0.f;
    for (int i = 0; i < 8; i++) { a += ps[i]; a2 += ps2[i]; }
    ps[0] = a * (1.f / C);
    ps2[0] = a2;
  }
  __syncthreads();
  float mu = ps[0];
  float var = ps2[0] * (1.f / C) - mu * mu;
  float rstd = rsqrtf(var + 1e-5f);
#pragma unroll
  for (int i = 0; i < 8; i++) {
    int c = threadIdx.x + i * 256;
    yr[c] = (v[i] - mu) * rstd * w[c] + b[c];
  }
}

// =====================================================================
// Time-mix: low-rank maa modulation + low-rank decay.  One block/token.
// Emits xk,xv,xr,xg (bf16 for WMMA GEMMs) and decay pre-activation w (f32).
// =====================================================================
__global__ __launch_bounds__(256) void tmix_kernel(
    const float* __restrict__ xn,
    const float* __restrict__ maa_x, const float* __restrict__ maa_w,
    const float* __restrict__ maa_k, const float* __restrict__ maa_v,
    const float* __restrict__ maa_r, const float* __restrict__ maa_g,
    const float* __restrict__ W1,   // C x 160
    const float* __restrict__ W2,   // 160 x C (5,32,C flattened)
    const float* __restrict__ tdecay,
    const float* __restrict__ DW1,  // C x 64
    const float* __restrict__ DW2,  // 64 x C
    __bf16* __restrict__ xk, __bf16* __restrict__ xv,
    __bf16* __restrict__ xr, __bf16* __restrict__ xg,
    float* __restrict__ wout) {
  __shared__ float sXN[C], sXX[C], sS[C];
  __shared__ float sY[160];
  int row = blockIdx.x;
  int t = row & (T - 1);
  size_t rb = (size_t)row * C;
  int tid = threadIdx.x;
#pragma unroll
  for (int i = 0; i < 8; i++) {
    int c = tid + i * 256;
    float a = xn[rb + c];
    float p = (t == 0) ? 0.f : xn[rb - C + c];
    float xx = p - a;
    sXN[c] = a; sXX[c] = xx;
    sS[c] = a + xx * maa_x[c];
  }
  __syncthreads();
  if (tid < 160) {
    float acc = 0.f;
    for (int c = 0; c < C; c++) acc += sS[c] * W1[(size_t)c * 160 + tid];
    sY[tid] = tanhf(acc);
  }
  __syncthreads();
#pragma unroll 1
  for (int i = 0; i < 8; i++) {
    int c = tid + i * 256;
    float m0 = 0, m1 = 0, m2 = 0, m3 = 0, m4 = 0;
#pragma unroll
    for (int e = 0; e < TM; e++) {
      m0 += sY[e]       * W2[(size_t)(e)       * C + c];
      m1 += sY[32 + e]  * W2[(size_t)(32 + e)  * C + c];
      m2 += sY[64 + e]  * W2[(size_t)(64 + e)  * C + c];
      m3 += sY[96 + e]  * W2[(size_t)(96 + e)  * C + c];
      m4 += sY[128 + e] * W2[(size_t)(128 + e) * C + c];
    }
    float a = sXN[c], xx = sXX[c];
    float xw = a + xx * (maa_w[c] + m0);
    xk[rb + c] = (__bf16)(a + xx * (maa_k[c] + m1));
    xv[rb + c] = (__bf16)(a + xx * (maa_v[c] + m2));
    xr[rb + c] = (__bf16)(a + xx * (maa_r[c] + m3));
    xg[rb + c] = (__bf16)(a + xx * (maa_g[c] + m4));
    sS[c] = xw;
  }
  __syncthreads();
  if (tid < TD) {
    float acc = 0.f;
    for (int c = 0; c < C; c++) acc += sS[c] * DW1[(size_t)c * TD + tid];
    sY[tid] = tanhf(acc);
  }
  __syncthreads();
#pragma unroll 1
  for (int i = 0; i < 8; i++) {
    int c = tid + i * 256;
    float acc = tdecay[c];
#pragma unroll
    for (int j = 0; j < TD; j++) acc += sY[j] * DW2[(size_t)j * C + c];
    wout[rb + c] = acc;
  }
}

// =====================================================================
// bf16 WMMA GEMM:  Out[M,N] = act(A[M,K] @ W[N,K]^T) (+ addsrc)
// BM=BN=128, BK=32, 256 threads = 8 waves (4 along M x 2 along N).
// Each wave owns a 32x64 slab of C: 2 A-frags x 4 B-frags = 8 WMMAs/k-step.
// LDS: 2 x (128x32 A + 128x32 W) bf16 double-buffered (40KB), filled with
// CDNA5 async global->LDS b128 transfers.
// =====================================================================
constexpr int LDK = 40;  // padded LDS row stride in halfs

__global__ __launch_bounds__(256) void gemm_bf16_kernel(
    const __bf16* __restrict__ A,   // M x K
    const __bf16* __restrict__ W,   // N x K
    float* __restrict__ Df,         // used when !out_bf16
    __bf16* __restrict__ Dbf,       // used when out_bf16
    const float* __restrict__ addsrc,  // nullable, M x N
    int M, int N, int K, int act, int out_bf16) {
  __shared__ __bf16 As[2][128 * LDK];
  __shared__ __bf16 Bs[2][128 * LDK];

  int tid = threadIdx.x;
  int wave = tid >> 5, lane = tid & 31;
  int wm = wave & 3, wn = wave >> 2;       // 4 x 2 wave grid
  int bm = blockIdx.y * 128, bn = blockIdx.x * 128;

  // loaders: each thread moves 2 x 16B segments per tile per matrix
  int lrow = tid >> 2, lseg = tid & 3;     // rows 0..63 (+64 for 2nd half)
  const __bf16* Abase = A + (size_t)(bm + lrow) * K + lseg * 8;
  const __bf16* Wbase = W + (size_t)(bn + lrow) * K + lseg * 8;
  size_t rowstride64 = (size_t)64 * K;
  int ldsoff = lrow * LDK + lseg * 8;

  auto load_tile = [&](int buf, int kt) {
    const __bf16* ga = Abase + (size_t)kt * 32;
    const __bf16* gw = Wbase + (size_t)kt * 32;
    __bf16* la = &As[buf][ldsoff];
    __bf16* lw = &Bs[buf][ldsoff];
#if HAVE_ASYNC_LDS
    __builtin_amdgcn_global_load_async_to_lds_b128((v4i*)ga, (v4i*)la, 0, 0);
    __builtin_amdgcn_global_load_async_to_lds_b128((v4i*)(ga + rowstride64),
                                                   (v4i*)(la + 64 * LDK), 0, 0);
    __builtin_amdgcn_global_load_async_to_lds_b128((v4i*)gw, (v4i*)lw, 0, 0);
    __builtin_amdgcn_global_load_async_to_lds_b128((v4i*)(gw + rowstride64),
                                                   (v4i*)(lw + 64 * LDK), 0, 0);
#else
    *(uint4*)la = *(const uint4*)ga;
    *(uint4*)(la + 64 * LDK) = *(const uint4*)(ga + rowstride64);
    *(uint4*)lw = *(const uint4*)gw;
    *(uint4*)(lw + 64 * LDK) = *(const uint4*)(gw + rowstride64);
#endif
  };

  f32x8 acc[2][4];
#pragma unroll
  for (int mi = 0; mi < 2; mi++)
#pragma unroll
    for (int ni = 0; ni < 4; ni++) acc[mi][ni] = (f32x8){};

  int nk = K / 32;

  load_tile(0, 0);
#if HAVE_ASYNC_LDS
  WAIT_ASYNC0();
#endif
  __syncthreads();

  for (int kt = 0; kt < nk; kt++) {
    int cur = kt & 1;
    if (kt + 1 < nk) load_tile(cur ^ 1, kt + 1);

    // A fragments: 16x32 bf16 each.  lane&15 = M row; K split by lane group.
    int ak = (lane & 16) ? 8 : 0;
    bf16x16 af[2];
#pragma unroll
    for (int mi = 0; mi < 2; mi++) {
      const __bf16* Ap = &As[cur][(wm * 32 + mi * 16 + (lane & 15)) * LDK];
#pragma unroll
      for (int i = 0; i < 8; i++) {
        af[mi][i]     = Ap[ak + i];
        af[mi][8 + i] = Ap[ak + 16 + i];
      }
    }
    // B fragments: 32x16 each; B[k][n] = Wtile[n][k]; lane&15 = N col.
    int bk = (lane & 16) ? 8 : 0;
#pragma unroll
    for (int ni = 0; ni < 4; ni++) {
      const __bf16* Bp = &Bs[cur][(wn * 64 + ni * 16 + (lane & 15)) * LDK];
      bf16x16 bfr;
#pragma unroll
      for (int hh = 0; hh < 8; hh++) { bfr[hh] = Bp[bk + hh]; bfr[8 + hh] = Bp[bk + 16 + hh]; }
      acc[0][ni] = __builtin_amdgcn_wmma_f32_16x16x32_bf16(
          false, af[0], false, bfr, (short)0, acc[0][ni], false, false);
      acc[1][ni] = __builtin_amdgcn_wmma_f32_16x16x32_bf16(
          false, af[1], false, bfr, (short)0, acc[1][ni], false, false);
    }
#if HAVE_ASYNC_LDS
    WAIT_ASYNC0();
#endif
    __syncthreads();
  }

  // Epilogue.  C/D frag: elem i -> M = i + 8*(lane>=16), N = lane&15.
  int ml = (lane & 16) ? 8 : 0;
  int nl = lane & 15;
#pragma unroll
  for (int mi = 0; mi < 2; mi++) {
#pragma unroll
    for (int i = 0; i < 8; i++) {
      int gm = bm + wm * 32 + mi * 16 + ml + i;
#pragma unroll
      for (int ni = 0; ni < 4; ni++) {
        int gn = bn + wn * 64 + ni * 16 + nl;
        float v = acc[mi][ni][i];
        if (act == 1)      v = v * (1.f / (1.f + __expf(-v)));         // silu
        else if (act == 2) v = 1.f / (1.f + __expf(-v));               // sigmoid
        else if (act == 3) { float r = v > 0.f ? v : 0.f; v = r * r; } // relu^2
        size_t di = (size_t)gm * N + gn;
        if (addsrc) v += addsrc[di];
        if (out_bf16) Dbf[di] = (__bf16)v;
        else          Df[di]  = v;
      }
    }
  }
}

// =====================================================================
// WKV6 recurrence.  One block per (b,h), 64 threads; lane j owns column j
// of the 64x64 state in registers.
// =====================================================================
__global__ __launch_bounds__(64) void wkv6_kernel(
    const float* __restrict__ r, const float* __restrict__ k,
    const float* __restrict__ v, const float* __restrict__ w,
    const float* __restrict__ u, float* __restrict__ out) {
  int bh = blockIdx.x;
  int b = bh / H, h = bh % H;
  int j = threadIdx.x;
  __shared__ float rr[HS], kk[HS], dd[HS], uu[HS];
  uu[j] = u[h * HS + j];
  float S[HS];
#pragma unroll
  for (int i = 0; i < HS; i++) S[i] = 0.f;
  size_t base = ((size_t)b * T) * C + (size_t)h * HS;
  for (int t = 0; t < T; t++) {
    size_t o = base + (size_t)t * C;
    __syncthreads();
    rr[j] = r[o + j];
    kk[j] = k[o + j];
    dd[j] = __expf(-__expf(w[o + j]));
    __syncthreads();
    float vj = v[o + j];
    float acc = 0.f;
#pragma unroll
    for (int i = 0; i < HS; i++) {
      float kv = kk[i] * vj;
      acc += rr[i] * fmaf(uu[i], kv, S[i]);
      S[i] = fmaf(S[i], dd[i], kv);
    }
    out[o + j] = acc;
  }
}

// =====================================================================
// GroupNorm over each head (64 elems) then * lnx_w + lnx_b, then * g.
// One block of 64 threads per (b,t,h).  Output bf16 for Wo GEMM.
// =====================================================================
__global__ __launch_bounds__(64) void gn_mul_kernel(
    const float* __restrict__ o, const float* __restrict__ g,
    const float* __restrict__ lw, const float* __restrict__ lb,
    __bf16* __restrict__ og) {
  int grp = blockIdx.x;
  int h = grp & (H - 1);
  size_t base = (size_t)grp * HS;
  int j = threadIdx.x;
  float val = o[base + j];
  __shared__ float part[2];
  float s = val;
#pragma unroll
  for (int off = 16; off > 0; off >>= 1) s += __shfl_down(s, off, 32);
  if ((j & 31) == 0) part[j >> 5] = s;
  __syncthreads();
  float mu = (part[0] + part[1]) * (1.f / 64.f);
  float d = val - mu;
  float d2 = d * d;
#pragma unroll
  for (int off = 16; off > 0; off >>= 1) d2 += __shfl_down(d2, off, 32);
  __syncthreads();
  if ((j & 31) == 0) part[j >> 5] = d2;
  __syncthreads();
  float var = (part[0] + part[1]) * (1.f / 64.f);
  float res = d * rsqrtf(var + 64e-5f);
  int c = h * HS + j;
  res = res * lw[c] + lb[c];
  og[base + j] = (__bf16)(res * g[base + j]);
}

// =====================================================================
// FFN token shift mix -> bf16 GEMM inputs.
// =====================================================================
__global__ __launch_bounds__(256) void ffnmix_kernel(
    const float* __restrict__ xn2, const float* __restrict__ mk,
    const float* __restrict__ mr, __bf16* __restrict__ xkf,
    __bf16* __restrict__ xrf) {
  int idx = blockIdx.x * 256 + threadIdx.x;
  int c = idx & (C - 1);
  int row = idx >> 11;  // /C
  int t = row & (T - 1);
  float xn = xn2[idx];
  float xp = (t == 0) ? 0.f : xn2[idx - C];
  float xx = xp - xn;
  xkf[idx] = (__bf16)(xn + xx * mk[c]);
  xrf[idx] = (__bf16)(xn + xx * mr[c]);
}

// =====================================================================
// Final combine: out = x1 + sigmoid_r * kv
// =====================================================================
__global__ __launch_bounds__(256) void final_kernel(
    const float* __restrict__ x1, const float* __restrict__ rs,
    const float* __restrict__ kv, float* __restrict__ out) {
  int idx = blockIdx.x * 256 + threadIdx.x;
  out[idx] = x1[idx] + rs[idx] * kv[idx];
}

// =====================================================================
// Host launcher
// =====================================================================
extern "C" void kernel_launch(void* const* d_in, const int* in_sizes, int n_in,
                              void* d_out, int out_size, void* d_ws, size_t ws_size,
                              hipStream_t stream) {
  (void)in_sizes; (void)n_in; (void)out_size; (void)ws_size;
  const float* x          = (const float*)d_in[0];
  const float* ln1_w      = (const float*)d_in[1];
  const float* ln1_b      = (const float*)d_in[2];
  const float* ln2_w      = (const float*)d_in[3];
  const float* ln2_b      = (const float*)d_in[4];
  const float* maa_x      = (const float*)d_in[5];
  const float* maa_w      = (const float*)d_in[6];
  const float* maa_k      = (const float*)d_in[7];
  const float* maa_v      = (const float*)d_in[8];
  const float* maa_r      = (const float*)d_in[9];
  const float* maa_g      = (const float*)d_in[10];
  const float* maa_w1     = (const float*)d_in[11];
  const float* maa_w2     = (const float*)d_in[12];
  const float* time_decay = (const float*)d_in[13];
  const float* decay_w1   = (const float*)d_in[14];
  const float* decay_w2   = (const float*)d_in[15];
  const float* time_faaaa = (const float*)d_in[16];
  const float* Wr         = (const float*)d_in[17];
  const float* Wk         = (const float*)d_in[18];
  const float* Wv         = (const float*)d_in[19];
  const float* Wg         = (const float*)d_in[20];
  const float* Wo         = (const float*)d_in[21];
  const float* lnx_w      = (const float*)d_in[22];
  const float* lnx_b      = (const float*)d_in[23];
  const float* ffn_maa_k  = (const float*)d_in[24];
  const float* ffn_maa_r  = (const float*)d_in[25];
  const float* Wk_ffn     = (const float*)d_in[26];
  const float* Wr_ffn     = (const float*)d_in[27];
  const float* Wv_ffn     = (const float*)d_in[28];
  float* out = (float*)d_out;

  // ---- workspace carve-up ----
  char* ws = (char*)d_ws;
  size_t off = 0;
  auto alloc = [&](size_t bytes) -> char* {
    char* p = ws + off;
    off += (bytes + 255) & ~(size_t)255;
    return p;
  };
  const size_t CC = (size_t)C * C;
  const size_t FC = (size_t)FFN * C;

  __bf16* wr_bf    = (__bf16*)alloc(CC * 2);
  __bf16* wk_bf    = (__bf16*)alloc(CC * 2);
  __bf16* wv_bf    = (__bf16*)alloc(CC * 2);
  __bf16* wg_bf    = (__bf16*)alloc(CC * 2);
  __bf16* wo_bf    = (__bf16*)alloc(CC * 2);
  __bf16* wrffn_bf = (__bf16*)alloc(CC * 2);
  __bf16* wkffn_bf = (__bf16*)alloc(FC * 2);
  __bf16* wvffn_bf = (__bf16*)alloc(FC * 2);

  float*  xn    = (float*)alloc((size_t)BTC * 4);
  __bf16* xk_bf = (__bf16*)alloc((size_t)BTC * 2);
  __bf16* xv_bf = (__bf16*)alloc((size_t)BTC * 2);
  __bf16* xr_bf = (__bf16*)alloc((size_t)BTC * 2);
  __bf16* xg_bf = (__bf16*)alloc((size_t)BTC * 2);
  float*  wdec  = (float*)alloc((size_t)BTC * 4);
  float*  r_    = (float*)alloc((size_t)BTC * 4);
  float*  k_    = (float*)alloc((size_t)BTC * 4);
  float*  v_    = (float*)alloc((size_t)BTC * 4);
  float*  g_    = (float*)alloc((size_t)BTC * 4);
  float*  o_    = (float*)alloc((size_t)BTC * 4);
  __bf16* og_bf = (__bf16*)alloc((size_t)BTC * 2);
  float*  x1    = (float*)alloc((size_t)BTC * 4);
  float*  xn2   = (float*)alloc((size_t)BTC * 4);
  __bf16* xkf_bf = (__bf16*)alloc((size_t)BTC * 2);
  __bf16* xrf_bf = (__bf16*)alloc((size_t)BTC * 2);
  __bf16* kf_bf  = (__bf16*)alloc((size_t)BT * FFN * 2);
  float*  kv_    = (float*)alloc((size_t)BTC * 4);
  float*  rs_    = (float*)alloc((size_t)BTC * 4);

  auto cvt = [&](const float* s, __bf16* d, size_t n) {
    cvt_bf16_kernel<<<(unsigned)((n + 255) / 256), 256, 0, stream>>>(s, d, (int)n);
  };
  cvt(Wr, wr_bf, CC);      cvt(Wk, wk_bf, CC);
  cvt(Wv, wv_bf, CC);      cvt(Wg, wg_bf, CC);
  cvt(Wo, wo_bf, CC);      cvt(Wr_ffn, wrffn_bf, CC);
  cvt(Wk_ffn, wkffn_bf, FC); cvt(Wv_ffn, wvffn_bf, FC);

  // ---- attention (time mix) ----
  ln_kernel<<<BT, 256, 0, stream>>>(x, ln1_w, ln1_b, xn);
  tmix_kernel<<<BT, 256, 0, stream>>>(xn, maa_x, maa_w, maa_k, maa_v, maa_r,
                                      maa_g, maa_w1, maa_w2, time_decay,
                                      decay_w1, decay_w2, xk_bf, xv_bf, xr_bf,
                                      xg_bf, wdec);

  auto gemm = [&](const __bf16* A, const __bf16* W, float* Df, __bf16* Dbf,
                  const float* add, int M, int N, int K, int act, int obf) {
    gemm_bf16_kernel<<<dim3(N / 128, M / 128), 256, 0, stream>>>(A, W, Df, Dbf,
                                                                 add, M, N, K,
                                                                 act, obf);
  };
  gemm(xr_bf, wr_bf, r_, nullptr, nullptr, BT, C, C, 0, 0);
  gemm(xk_bf, wk_bf, k_, nullptr, nullptr, BT, C, C, 0, 0);
  gemm(xv_bf, wv_bf, v_, nullptr, nullptr, BT, C, C, 0, 0);
  gemm(xg_bf, wg_bf, g_, nullptr, nullptr, BT, C, C, 1, 0);  // silu

  wkv6_kernel<<<B * H, HS, 0, stream>>>(r_, k_, v_, wdec, time_faaaa, o_);
  gn_mul_kernel<<<BT * H, HS, 0, stream>>>(o_, g_, lnx_w, lnx_b, og_bf);

  gemm(og_bf, wo_bf, x1, nullptr, x, BT, C, C, 0, 0);  // residual add

  // ---- channel mix ----
  ln_kernel<<<BT, 256, 0, stream>>>(x1, ln2_w, ln2_b, xn2);
  ffnmix_kernel<<<BTC / 256, 256, 0, stream>>>(xn2, ffn_maa_k, ffn_maa_r,
                                               xkf_bf, xrf_bf);
  gemm(xkf_bf, wkffn_bf, nullptr, kf_bf, nullptr, BT, FFN, C, 3, 1);  // relu^2
  gemm(kf_bf, wvffn_bf, kv_, nullptr, nullptr, BT, C, FFN, 0, 0);
  gemm(xrf_bf, wrffn_bf, rs_, nullptr, nullptr, BT, C, C, 2, 0);      // sigmoid

  final_kernel<<<BTC / 256, 256, 0, stream>>>(x1, rs_, kv_, out);
}
